// DMA_68332929679604
// MI455X (gfx1250) — compile-verified, matrix operand order checked by reading, not run on hardware
//
#include <hip/hip_runtime.h>

typedef __attribute__((ext_vector_type(2)))  float  v2f;
typedef __attribute__((ext_vector_type(4)))  float  v4f;
typedef __attribute__((ext_vector_type(8)))  float  v8f;
typedef __attribute__((ext_vector_type(16))) __bf16 v16bf;
typedef __attribute__((ext_vector_type(8)))  unsigned short v8us;
typedef __attribute__((ext_vector_type(16))) unsigned short v16us;

#define B_    4
#define CK_   64
#define CV_   512
#define THW_  8000
#define HW_   1600
#define MC_   32      // memory rows per chunk (K of readout WMMA)
#define NQ_   16      // queries per workgroup (one WMMA N tile)
#define SCALE_ 0.125f // 1/sqrt(CK)
#define DOTPITCH 17   // padded pitch for dot tile (bank-conflict-free)
#define PPITCH   36   // padded pitch (ushorts) for bf16 P rows

static __device__ __forceinline__ __bf16 f2bf(float f) { return (__bf16)f; }

// ---------------------------------------------------------------------------
// Prologue 1: mem_val fp32 -> bf16 (done ONCE; hot loop then streams bf16 from
// L2 at half the bandwidth with zero conversion VALU).
// ---------------------------------------------------------------------------
__global__ __launch_bounds__(256) void cvt_mv_bf16(const float* __restrict__ src,
                                                   unsigned short* __restrict__ dst,
                                                   int n8) {
  int i = blockIdx.x * 256 + threadIdx.x;
  if (i >= n8) return;
  const v4f* s = (const v4f*)src + (size_t)i * 2;
  v4f a = s[0], b = s[1];
  v8us o;
#pragma unroll
  for (int j = 0; j < 4; ++j) {
    o[j]     = __builtin_bit_cast(unsigned short, f2bf(a[j]));
    o[4 + j] = __builtin_bit_cast(unsigned short, f2bf(b[j]));
  }
  *((v8us*)dst + i) = o;
}

// ---------------------------------------------------------------------------
// Prologue 2: ||mk||^2 per memory location (coalesced across lanes per channel).
// ---------------------------------------------------------------------------
__global__ __launch_bounds__(256) void mksq_kernel(const float* __restrict__ mk,
                                                   float* __restrict__ mksq) {
  int gid = blockIdx.x * 256 + threadIdx.x;  // 0 .. B*THW-1
  if (gid >= B_ * THW_) return;
  int b = gid / THW_;
  int m = gid - b * THW_;
  const float* p = mk + (size_t)b * CK_ * THW_ + m;
  float s = 0.f;
#pragma unroll 8
  for (int c = 0; c < CK_; ++c) {
    float v = p[(size_t)c * THW_];
    s += v * v;
  }
  mksq[gid] = s;
}

// ---------------------------------------------------------------------------
// Fused STCN memory read: per workgroup = one 16-query tile, flash-style pass
// over THW in chunks of 32. fp32 WMMA logits, fp32 online softmax, bf16 WMMA
// readout with fp32 accumulation.
// ---------------------------------------------------------------------------
template <bool WS>
__global__ __launch_bounds__(256, 1) void stcn_mem_read(
    const float* __restrict__ mem_key,
    const float* __restrict__ mem_val,            // fallback path
    const unsigned short* __restrict__ mv_bf,     // WS path: bf16 mem_val
    const float* __restrict__ mksq_pre,           // WS path: precomputed ||mk||^2
    const float* __restrict__ qry_key,
    float* __restrict__ out)
{
  __shared__ float          lds_qk[CK_ * NQ_];        // [c][n] query-key tile
  __shared__ float          lds_dot[MC_ * DOTPITCH];  // [m][n] raw dot products
  __shared__ float          lds_mksq[MC_];            // ||mk||^2 per chunk row
  __shared__ unsigned short lds_p[NQ_ * PPITCH];      // [n][k] bf16 probabilities
  __shared__ float          lds_rs[NQ_];              // accumulator rescale per query
  __shared__ float          lds_invl[NQ_];            // 1/sum per query

  const int tid  = threadIdx.x;
  const int wave = tid >> 5;
  const int lane = tid & 31;
  const int n    = lane & 15;   // WMMA column (query) for B/C/D layouts
  const int hi   = lane >> 4;   // half-wave selector
  const int b    = blockIdx.y;
  const int q0   = blockIdx.x * NQ_;

  const float* mk = mem_key + (size_t)b * CK_ * THW_;
  const float* qk = qry_key + (size_t)b * CK_ * HW_;

  // Stage qk tile [CK][NQ] into LDS
  for (int i = tid; i < CK_ * NQ_; i += 256) {
    int c = i >> 4, nn = i & 15;
    lds_qk[i] = qk[(size_t)c * HW_ + q0 + nn];
  }
  __syncthreads();

  // Output accumulators: each wave owns 4 CV tiles (64 rows of CV).
  v8f acc[4];
#pragma unroll
  for (int t = 0; t < 4; ++t)
#pragma unroll
    for (int g = 0; g < 8; ++g) acc[t][g] = 0.f;

  float m_run = -INFINITY;  // running softmax max (wave0 lanes 0..15 only)
  float l_run = 0.f;        // running softmax sum
  const int cv_base = wave * 64;

  for (int m0 = 0; m0 < THW_; m0 += MC_) {
    // ---- Stage 0: all waves issue mv chunk loads as bf16 A-fragments.
    // 16-bit A layout: VGPR e(0..3): K=2e+8*hi; e(4..7): K=16+2(e-4)+8*hi
    // -> lane needs elements [8*hi..8*hi+7] and [16+8*hi..23+8*hi] of the chunk.
    v16bf afr[4];
#pragma unroll
    for (int t = 0; t < 4; ++t) {
      if constexpr (WS) {
        const unsigned short* src =
            mv_bf + ((size_t)b * CV_ + cv_base + 16 * t + n) * THW_ + m0 + 8 * hi;
        v8us h0 = *(const v8us*)(src);        // one global_load_b128
        v8us h1 = *(const v8us*)(src + 16);   // one global_load_b128
        v16us u;
#pragma unroll
        for (int j = 0; j < 8; ++j) { u[j] = h0[j]; u[8 + j] = h1[j]; }
        afr[t] = __builtin_bit_cast(v16bf, u);
      } else {
        const float* src =
            mem_val + ((size_t)b * CV_ + cv_base + 16 * t + n) * THW_ + m0 + 8 * hi;
        v4f s0 = *(const v4f*)(src + 0);
        v4f s1 = *(const v4f*)(src + 4);
        v4f s2 = *(const v4f*)(src + 16);
        v4f s3 = *(const v4f*)(src + 20);
#pragma unroll
        for (int j = 0; j < 4; ++j) {
          afr[t][j]      = f2bf(s0[j]);
          afr[t][4 + j]  = f2bf(s1[j]);
          afr[t][8 + j]  = f2bf(s2[j]);
          afr[t][12 + j] = f2bf(s3[j]);
        }
      }
    }

    // ---- Stage 1: waves 0/1: two 16x16 logit tiles (fp32 WMMA, K=64);
    //      wave 2: ||mk||^2 for the 32 chunk rows.
    if (wave < 2) {
      v8f d;
#pragma unroll
      for (int g = 0; g < 8; ++g) d[g] = 0.f;
      const size_t mrow = (size_t)(m0 + wave * 16 + n);
#pragma unroll
      for (int kk = 0; kk < CK_; kk += 4) {
        // f32 A 16x4: VGPR0 K=0(+2*hi), VGPR1 K=1(+2*hi); B mirrors, N on lanes.
        v2f a, bb;
        a.x  = mk[(size_t)(kk + 2 * hi) * THW_ + mrow];
        a.y  = mk[(size_t)(kk + 1 + 2 * hi) * THW_ + mrow];
        bb.x = lds_qk[(kk + 2 * hi) * NQ_ + n];
        bb.y = lds_qk[(kk + 1 + 2 * hi) * NQ_ + n];
        d = __builtin_amdgcn_wmma_f32_16x16x4_f32(false, a, false, bb,
                                                  (short)0, d, false, false);
      }
      // C/D layout: VGPR g -> row g + 8*hi, lane -> col n
#pragma unroll
      for (int g = 0; g < 8; ++g)
        lds_dot[(wave * 16 + g + 8 * hi) * DOTPITCH + n] = d[g];
    } else if (wave == 2) {
      if constexpr (WS) {
        lds_mksq[lane] = mksq_pre[(size_t)b * THW_ + m0 + lane];
      } else {
        float ssum = 0.f;
        const size_t mrow = (size_t)(m0 + lane);
#pragma unroll 8
        for (int c = 0; c < CK_; ++c) {
          float v = mk[(size_t)c * THW_ + mrow];
          ssum += v * v;
        }
        lds_mksq[lane] = ssum;
      }
    }
    __syncthreads();

    // ---- Stage 2: online softmax for the chunk (16 lanes, one per query col).
    if (tid < 16) {
      float cmax = -INFINITY;
#pragma unroll 8
      for (int r = 0; r < MC_; ++r) {
        float s = (2.f * lds_dot[r * DOTPITCH + tid] - lds_mksq[r]) * SCALE_;
        cmax = fmaxf(cmax, s);
      }
      float mnew = fmaxf(m_run, cmax);
      float rs   = __expf(m_run - mnew);  // first chunk: exp(-inf) = 0
      float csum = 0.f;
#pragma unroll 4
      for (int r = 0; r < MC_; r += 2) {
        float s0 = (2.f * lds_dot[r * DOTPITCH + tid] - lds_mksq[r]) * SCALE_ - mnew;
        float s1 = (2.f * lds_dot[(r + 1) * DOTPITCH + tid] - lds_mksq[r + 1]) * SCALE_ - mnew;
        float p0 = __expf(s0), p1 = __expf(s1);
        csum += p0 + p1;
        unsigned pack = (unsigned)__builtin_bit_cast(unsigned short, f2bf(p0))
                      | ((unsigned)__builtin_bit_cast(unsigned short, f2bf(p1)) << 16);
        *(unsigned*)&lds_p[tid * PPITCH + r] = pack;  // row=query, packed bf16 pairs
      }
      l_run = l_run * rs + csum;
      m_run = mnew;
      lds_rs[tid] = rs;
    }
    __syncthreads();

    // ---- Stage 3: rescale accumulators, build bf16 B-frag of P, 4 bf16 WMMAs.
    float r = lds_rs[n];
#pragma unroll
    for (int t = 0; t < 4; ++t)
#pragma unroll
      for (int g = 0; g < 8; ++g) acc[t][g] *= r;

    // 16-bit B layout mirrors A: lane col n; e(0..3): K=2e+8*hi; e(4..7): K=16+2(e-4)+8*hi
    v16bf bfr;
    const unsigned* prow = (const unsigned*)(lds_p + n * PPITCH);
#pragma unroll
    for (int e = 0; e < 4; ++e) {
      unsigned pp = prow[e + 4 * hi];
      bfr[2 * e]     = __builtin_bit_cast(__bf16, (unsigned short)(pp & 0xFFFFu));
      bfr[2 * e + 1] = __builtin_bit_cast(__bf16, (unsigned short)(pp >> 16));
    }
#pragma unroll
    for (int e = 0; e < 4; ++e) {
      unsigned pp = prow[8 + e + 4 * hi];
      bfr[8 + 2 * e] = __builtin_bit_cast(__bf16, (unsigned short)(pp & 0xFFFFu));
      bfr[9 + 2 * e] = __builtin_bit_cast(__bf16, (unsigned short)(pp >> 16));
    }

#pragma unroll
    for (int t = 0; t < 4; ++t)
      acc[t] = __builtin_amdgcn_wmma_f32_16x16x32_bf16(false, afr[t], false, bfr,
                                                       (short)0, acc[t], false, false);
    __syncthreads();  // protect LDS (dot/P) before next chunk overwrites
  }

  // ---- Finalize: divide by softmax sum and store.
  if (tid < 16) lds_invl[tid] = 1.f / l_run;
  __syncthreads();
  float invl = lds_invl[n];
#pragma unroll
  for (int t = 0; t < 4; ++t) {
#pragma unroll
    for (int g = 0; g < 8; ++g) {
      int cv = cv_base + 16 * t + g + 8 * hi;
      out[((size_t)b * CV_ + cv) * HW_ + q0 + n] = acc[t][g] * invl;
    }
  }
}

extern "C" void kernel_launch(void* const* d_in, const int* in_sizes, int n_in,
                              void* d_out, int out_size, void* d_ws, size_t ws_size,
                              hipStream_t stream) {
  (void)in_sizes; (void)n_in; (void)out_size;
  const float* mem_key = (const float*)d_in[0];
  const float* mem_val = (const float*)d_in[1];
  const float* qry_key = (const float*)d_in[2];
  float* out = (float*)d_out;

  dim3 grid(HW_ / NQ_, B_);  // 100 query tiles x 4 batches

  const size_t bf_bytes = (size_t)B_ * CV_ * THW_ * sizeof(unsigned short); // 32.8 MB
  const size_t sq_bytes = (size_t)B_ * THW_ * sizeof(float);                // 128 KB

  if (d_ws != nullptr && ws_size >= bf_bytes + sq_bytes) {
    unsigned short* mv_bf = (unsigned short*)d_ws;
    float* mksq = (float*)((char*)d_ws + bf_bytes);
    const int n8 = B_ * CV_ * THW_ / 8;
    cvt_mv_bf16<<<(n8 + 255) / 256, 256, 0, stream>>>(mem_val, mv_bf, n8);
    mksq_kernel<<<(B_ * THW_ + 255) / 256, 256, 0, stream>>>(mem_key, mksq);
    stcn_mem_read<true><<<grid, 256, 0, stream>>>(mem_key, mem_val, mv_bf, mksq,
                                                  qry_key, out);
  } else {
    stcn_mem_read<false><<<grid, 256, 0, stream>>>(mem_key, mem_val, nullptr,
                                                   nullptr, qry_key, out);
  }
}
